// MultiHeadedAttention_26680336843266
// MI455X (gfx1250) — matrix operand order repaired and checked by reading, not compiled, not run on hardware
//
#include <hip/hip_runtime.h>
#include <hip/hip_bf16.h>

// ---------------------------------------------------------------------------
// MI455X (gfx1250) relative-position multi-head attention.
// B=16, L=512, DM=512, H=8, DK=64, NREL=4, VS=5.
// All GEMMs via v_wmma_f32_16x16x32_bf16 (wave32), double-buffered fragment
// loads, weight tiles staged in LDS (bf16), 64x512 score block in LDS.
// ---------------------------------------------------------------------------

typedef __bf16 v16bf __attribute__((ext_vector_type(16)));
typedef float  v8f   __attribute__((ext_vector_type(8)));

union AF { v16bf v; __bf16 e[16]; uint4 q[2]; };

static __device__ inline v8f vzero8() {
  v8f z;
#pragma unroll
  for (int i = 0; i < 8; ++i) z[i] = 0.0f;
  return z;
}

static __device__ inline void cvt4(__bf16* d, float4 s) {
  d[0] = (__bf16)s.x; d[1] = (__bf16)s.y; d[2] = (__bf16)s.z; d[3] = (__bf16)s.w;
}

static __device__ inline float wave_max(float v) {
#pragma unroll
  for (int off = 16; off >= 1; off >>= 1) v = fmaxf(v, __shfl_xor(v, off, 32));
  return v;
}
static __device__ inline float wave_sum(float v) {
#pragma unroll
  for (int off = 16; off >= 1; off >>= 1) v += __shfl_xor(v, off, 32);
  return v;
}

#define B_  16
#define L_  512
#define DM_ 512
#define H_  8
#define DK_ 64
#define NREL_ 4
#define VS_ 5
#define WPAD_ (DM_ + 8)   // bf16 row pad: stride 1040B (16B-aligned, bank-spread)

// ---------------------------------------------------------------------------
// Kernel 1: projection GEMM  out = X @ W^T + bias   (M=8192, N=512, K=512)
// mode 0: Q -> bf16 (b,h,t,dk) pre-scaled by 0.125
// mode 1: K -> bf16 (b,h,t,dk)
// mode 2: V -> bf16 (b,h,dk,t)  (transposed for contiguous P@V B-frags)
// 128 thr = 4 waves; wave = 16(M)x64(N) tile; W tile staged once in LDS.
// ---------------------------------------------------------------------------
__global__ __launch_bounds__(128) void proj_kernel(
    const float* __restrict__ X, const float* __restrict__ W,
    const float* __restrict__ bias, __bf16* __restrict__ dst, int mode) {
  __shared__ __bf16 WB[64][WPAD_];   // 65 KB: W[n0..n0+63][:] as bf16

  const int lane = threadIdx.x & 31;
  const int wave = threadIdx.x >> 5;
  const int row0 = (blockIdx.x * 4 + wave) * 16;
  const int n0 = blockIdx.y * 64;
  const int m = lane & 15, hi = lane >> 4;

  // ---- cooperative stage of W tile (fp32 -> bf16) ----
  {
    const float4* w4 = (const float4*)(W + (size_t)n0 * DM_);
    for (int idx = threadIdx.x; idx < 64 * 128; idx += 128) {
      const int nl = idx >> 7, k4 = idx & 127;
      cvt4(&WB[nl][k4 * 4], w4[(size_t)nl * 128 + k4]);
    }
  }
  __syncthreads();

  v8f acc[4];
#pragma unroll
  for (int t = 0; t < 4; ++t) acc[t] = vzero8();

  const float* arow = X + (size_t)(row0 + m) * DM_;

  // double-buffered raw A (2 chunks of 8 floats each)
  float4 ar[2][4];
  {
    const float4* p = (const float4*)(arow + 8 * hi);
    ar[0][0] = p[0]; ar[0][1] = p[1]; ar[0][2] = p[4]; ar[0][3] = p[5];
  }
  for (int kb = 0; kb < DM_; kb += 32) {
    const int cur = (kb >> 5) & 1, nxt = cur ^ 1;
    if (kb + 32 < DM_) {  // prefetch next k-step before consuming current
      const float4* p = (const float4*)(arow + kb + 32 + 8 * hi);
      ar[nxt][0] = p[0]; ar[nxt][1] = p[1]; ar[nxt][2] = p[4]; ar[nxt][3] = p[5];
    }
    AF a;
    cvt4(&a.e[0], ar[cur][0]); cvt4(&a.e[4], ar[cur][1]);
    cvt4(&a.e[8], ar[cur][2]); cvt4(&a.e[12], ar[cur][3]);
#pragma unroll
    for (int nt = 0; nt < 4; ++nt) {
      AF bf;
      const uint4* ws = (const uint4*)&WB[nt * 16 + m][kb + 8 * hi];
      bf.q[0] = ws[0]; bf.q[1] = ws[2];
      acc[nt] = __builtin_amdgcn_wmma_f32_16x16x32_bf16(
          false, a.v, false, bf.v, (short)0, acc[nt], false, false);
    }
  }

#pragma unroll
  for (int nt = 0; nt < 4; ++nt) {
    const int n = n0 + nt * 16 + m;
    const float bv = bias[n];
    const int h = n >> 6, dk = n & 63;
#pragma unroll
    for (int r = 0; r < 8; ++r) {
      const int mm = row0 + r + 8 * hi;
      const int b = mm >> 9, t = mm & 511;
      const float val = acc[nt][r] + bv;
      if (mode == 2) {
        dst[((size_t)(b * H_ + h) * DK_ + dk) * L_ + t] = (__bf16)val;
      } else {
        const float s = (mode == 0) ? val * 0.125f : val;
        dst[((size_t)(b * H_ + h) * L_ + t) * DK_ + dk] = (__bf16)s;
      }
    }
  }
}

// ---------------------------------------------------------------------------
// Kernel 2: attention. One block = one (b,h) and one 64-query tile.
// 128 threads = 4 waves; wave w owns local query rows [16w, 16w+16).
// Full 64x512 score block in LDS (128 KB of the 320 KB WGP LDS).
// ---------------------------------------------------------------------------
__global__ __launch_bounds__(128) void attn_kernel(
    const __bf16* __restrict__ Qws, const __bf16* __restrict__ Kws,
    const __bf16* __restrict__ Vtws, const int* __restrict__ rel_mat,
    const unsigned char* __restrict__ mask,
    const float* __restrict__ score_emb, const float* __restrict__ value_emb,
    __bf16* __restrict__ Xws) {
  __shared__ float Smat[64 * L_];            // 128 KB
  __shared__ float rpsL[NREL_][64][VS_];     // 5 KB
  __shared__ float acmL[64][16];             // 4 KB
  __shared__ float rowscale[64];
  __shared__ float vemb[NREL_][4][DK_];      // 4 KB

  const int bh = blockIdx.x;
  const int b = bh >> 3, h = bh & 7;
  const int q0 = blockIdx.y * 64;
  const int lane = threadIdx.x & 31;
  const int wave = threadIdx.x >> 5;
  const int m = lane & 15, hi = lane >> 4;

  const __bf16* Qbh = Qws + (size_t)bh * L_ * DK_;
  const __bf16* Kbh = Kws + (size_t)bh * L_ * DK_;
  const __bf16* Vbh = Vtws + (size_t)bh * DK_ * L_;

  for (int idx = threadIdx.x; idx < NREL_ * 4 * DK_; idx += 128) {
    const int r = idx / (4 * DK_);
    const int s = (idx / DK_) & 3;
    const int d = idx & 63;
    vemb[r][s][d] = value_emb[(((size_t)r * H_ + h) * VS_ + (s + 1)) * DK_ + d];
  }

  for (int idx = threadIdx.x; idx < NREL_ * 64 * VS_; idx += 128) {
    const int r = idx / (64 * VS_);
    const int rem = idx % (64 * VS_);
    const int il = rem / VS_, v = rem % VS_;
    const __bf16* qp = Qbh + (size_t)(q0 + il) * DK_;
    const float* ep = score_emb + (((size_t)r * H_ + h) * VS_ + v) * DK_;
    float s = 0.f;
#pragma unroll 8
    for (int d = 0; d < DK_; ++d) s += (float)qp[d] * ep[d];
    rpsL[r][il][v] = s;
  }
  __syncthreads();

  const int il0 = wave * 16;

  // ---- S = (scaled Q) K^T + rel_score, masked -> LDS ----
  AF qa[2];
#pragma unroll
  for (int ks = 0; ks < 2; ++ks) {
    const uint4* p = (const uint4*)(Qbh + (size_t)(q0 + il0 + m) * DK_ + ks * 32 + 16 * hi);
    qa[ks].q[0] = p[0];   // d +0..7
    qa[ks].q[1] = p[1];   // d +8..15
  }

  AF kf[2][2];  // double-buffered K fragments [buf][ks]
#pragma unroll
  for (int ks = 0; ks < 2; ++ks) {
    const uint4* p = (const uint4*)(Kbh + (size_t)m * DK_ + ks * 32 + 16 * hi);
    kf[0][ks].q[0] = p[0]; kf[0][ks].q[1] = p[1];
  }

  for (int jt = 0; jt < 32; ++jt) {
    const int cur = jt & 1, nxt = cur ^ 1;
    if (jt + 1 < 32) {
#pragma unroll
      for (int ks = 0; ks < 2; ++ks) {
        const uint4* p = (const uint4*)(Kbh + (size_t)((jt + 1) * 16 + m) * DK_ + ks * 32 + 16 * hi);
        kf[nxt][ks].q[0] = p[0]; kf[nxt][ks].q[1] = p[1];
      }
    }
    v8f acc = vzero8();
#pragma unroll
    for (int ks = 0; ks < 2; ++ks)
      acc = __builtin_amdgcn_wmma_f32_16x16x32_bf16(
          false, qa[ks].v, false, kf[cur][ks].v, (short)0, acc, false, false);

    const int j = jt * 16 + m;
#pragma unroll
    for (int r = 0; r < 8; ++r) {
      const int il = il0 + r + 8 * hi;
      const int ig = q0 + il;
      float s = acc[r];
#pragma unroll
      for (int rr = 0; rr < NREL_; ++rr) {
        const int id = rel_mat[(((size_t)b * NREL_ + rr) * L_ + ig) * L_ + j];
        s += rpsL[rr][il][id];
      }
      if (!mask[((size_t)b * L_ + ig) * L_ + j]) s = -1e9f;
      Smat[il * L_ + j] = s;
    }
  }
  __syncthreads();

  // ---- per-row softmax (unnormalized exp in LDS, 1/sum in rowscale) ----
  for (int rr = 0; rr < 16; ++rr) {
    const int il = il0 + rr;
    float* row = Smat + il * L_;
    float mx = -1e30f;
    for (int t = lane; t < L_; t += 32) mx = fmaxf(mx, row[t]);
    mx = wave_max(mx);
    float sum = 0.f;
    for (int t = lane; t < L_; t += 32) {
      const float e = __expf(row[t] - mx);
      row[t] = e;
      sum += e;
    }
    sum = wave_sum(sum);
    if (lane == 0) rowscale[il] = 1.0f / sum;
  }
  __syncthreads();

  // ---- acm buckets ----
  for (int rr = 0; rr < 16; ++rr) {
    const int il = il0 + rr;
    const int ig = q0 + il;
    float acc16[16];
#pragma unroll
    for (int t = 0; t < 16; ++t) acc16[t] = 0.f;
    const float* row = Smat + il * L_;
    for (int t = lane; t < L_; t += 32) {
      const float p = row[t];
#pragma unroll
      for (int r4 = 0; r4 < NREL_; ++r4) {
        const int id = rel_mat[(((size_t)b * NREL_ + r4) * L_ + ig) * L_ + t];
        if (id > 0) acc16[r4 * 4 + (id - 1)] += p;
      }
    }
#pragma unroll
    for (int t = 0; t < 16; ++t) acc16[t] = wave_sum(acc16[t]);
    if (lane == 0) {
      const float sc = rowscale[il];
#pragma unroll
      for (int t = 0; t < 16; ++t) acmL[il][t] = acc16[t] * sc;
    }
  }
  __syncthreads();

  // ---- x = P @ V (+ acm . value_emb) -> bf16 ws ----
  const float rs = rowscale[il0 + m];
  for (int dt = 0; dt < 4; ++dt) {
    v8f acc = vzero8();
    AF vf[2];  // double-buffered V fragments
    {
      const uint4* p = (const uint4*)(Vbh + (size_t)(dt * 16 + m) * L_ + 16 * hi);
      vf[0].q[0] = p[0]; vf[0].q[1] = p[1];
    }
    for (int jb = 0; jb < L_; jb += 32) {
      const int cur = (jb >> 5) & 1, nxt = cur ^ 1;
      if (jb + 32 < L_) {
        const uint4* p = (const uint4*)(Vbh + (size_t)(dt * 16 + m) * L_ + jb + 32 + 16 * hi);
        vf[nxt].q[0] = p[0]; vf[nxt].q[1] = p[1];
      }
      AF pa;
      const float* prow = Smat + (il0 + m) * L_ + jb + 16 * hi;
#pragma unroll
      for (int e = 0; e < 8; ++e) {
        pa.e[e]     = (__bf16)(prow[e] * rs);
        pa.e[8 + e] = (__bf16)(prow[8 + e] * rs);
      }
      acc = __builtin_amdgcn_wmma_f32_16x16x32_bf16(
          false, pa.v, false, vf[cur].v, (short)0, acc, false, false);
    }
    const int dk = dt * 16 + m;
#pragma unroll
    for (int r = 0; r < 8; ++r) {
      const int il = il0 + r + 8 * hi;
      const int ig = q0 + il;
      float x = acc[r];
#pragma unroll
      for (int r4 = 0; r4 < NREL_; ++r4)
#pragma unroll
        for (int s = 0; s < 4; ++s)
          x += acmL[il][r4 * 4 + s] * vemb[r4][s][dk];
      Xws[((size_t)b * L_ + ig) * DM_ + h * DK_ + dk] = (__bf16)x;
    }
  }
}

// ---------------------------------------------------------------------------
// Kernel 3: output projection  out = X @ Wo^T + bo  (bf16 A, fp32 out)
// ---------------------------------------------------------------------------
__global__ __launch_bounds__(128) void out_proj_kernel(
    const __bf16* __restrict__ Xb, const float* __restrict__ W,
    const float* __restrict__ bias, float* __restrict__ out) {
  __shared__ __bf16 WB[64][WPAD_];

  const int lane = threadIdx.x & 31;
  const int wave = threadIdx.x >> 5;
  const int row0 = (blockIdx.x * 4 + wave) * 16;
  const int n0 = blockIdx.y * 64;
  const int m = lane & 15, hi = lane >> 4;

  {
    const float4* w4 = (const float4*)(W + (size_t)n0 * DM_);
    for (int idx = threadIdx.x; idx < 64 * 128; idx += 128) {
      const int nl = idx >> 7, k4 = idx & 127;
      cvt4(&WB[nl][k4 * 4], w4[(size_t)nl * 128 + k4]);
    }
  }
  __syncthreads();

  v8f acc[4];
#pragma unroll
  for (int t = 0; t < 4; ++t) acc[t] = vzero8();

  const __bf16* arow = Xb + (size_t)(row0 + m) * DM_;
  uint4 ar[2][2];
  {
    const uint4* p = (const uint4*)(arow + 8 * hi);
    ar[0][0] = p[0]; ar[0][1] = p[2];
  }
  for (int kb = 0; kb < DM_; kb += 32) {
    const int cur = (kb >> 5) & 1, nxt = cur ^ 1;
    if (kb + 32 < DM_) {
      const uint4* p = (const uint4*)(arow + kb + 32 + 8 * hi);
      ar[nxt][0] = p[0]; ar[nxt][1] = p[2];
    }
    AF a;
    a.q[0] = ar[cur][0]; a.q[1] = ar[cur][1];
#pragma unroll
    for (int nt = 0; nt < 4; ++nt) {
      AF bf;
      const uint4* ws = (const uint4*)&WB[nt * 16 + m][kb + 8 * hi];
      bf.q[0] = ws[0]; bf.q[1] = ws[2];
      acc[nt] = __builtin_amdgcn_wmma_f32_16x16x32_bf16(
          false, a.v, false, bf.v, (short)0, acc[nt], false, false);
    }
  }
#pragma unroll
  for (int nt = 0; nt < 4; ++nt) {
    const int n = n0 + nt * 16 + m;
    const float bv = bias[n];
#pragma unroll
    for (int r = 0; r < 8; ++r) {
      const int mm = row0 + r + 8 * hi;
      out[(size_t)mm * DM_ + n] = acc[nt][r] + bv;
    }
  }
}

// ---------------------------------------------------------------------------
extern "C" void kernel_launch(void* const* d_in, const int* in_sizes, int n_in,
                              void* d_out, int out_size, void* d_ws, size_t ws_size,
                              hipStream_t stream) {
  const float* query = (const float*)d_in[0];
  const float* key   = (const float*)d_in[1];
  const float* value = (const float*)d_in[2];
  const int*   rel   = (const int*)d_in[3];
  const unsigned char* mask = (const unsigned char*)d_in[4];
  const float* Wq = (const float*)d_in[5];
  const float* bq = (const float*)d_in[6];
  const float* Wk = (const float*)d_in[7];
  const float* bk = (const float*)d_in[8];
  const float* Wv = (const float*)d_in[9];
  const float* bv = (const float*)d_in[10];
  const float* Wo = (const float*)d_in[11];
  const float* bo = (const float*)d_in[12];
  const float* score_emb = (const float*)d_in[13];
  const float* value_emb = (const float*)d_in[14];
  float* out = (float*)d_out;

  const size_t elems = (size_t)B_ * L_ * DM_;
  __bf16* Qws  = (__bf16*)d_ws;
  __bf16* Kws  = Qws + elems;
  __bf16* Vtws = Kws + elems;
  __bf16* Xws  = Vtws + elems;

  dim3 gGemm(128, 8);
  dim3 blk(128);
  proj_kernel<<<gGemm, blk, 0, stream>>>(query, Wq, bq, Qws, 0);
  proj_kernel<<<gGemm, blk, 0, stream>>>(key,   Wk, bk, Kws, 1);
  proj_kernel<<<gGemm, blk, 0, stream>>>(value, Wv, bv, Vtws, 2);

  dim3 gAttn(B_ * H_, L_ / 64);
  attn_kernel<<<gAttn, blk, 0, stream>>>(Qws, Kws, Vtws, rel, mask,
                                         score_emb, value_emb, Xws);

  out_proj_kernel<<<gGemm, blk, 0, stream>>>(Xws, Wo, bo, out);
}
